// TransformerCrossEncoder_38646115729752
// MI455X (gfx1250) — matrix operand order repaired and checked by reading, not compiled, not run on hardware
//
#include <hip/hip_runtime.h>
#include <hip/hip_bf16.h>
#include <stdint.h>

// ---------------------------------------------------------------------------
// CDNA5 (gfx1250) transformer cross-encoder forward.
// - all matmuls on v_wmma_f32_16x16x32_bf16
// - GEMM tiles staged LDS via Tensor Data Mover (tensor_load_to_lds),
//   double-buffered with s_wait_tensorcnt
// - weights converted to bf16 once per launch; activations produced in bf16
// L=2 layers, M=2 streams, E=512, H=8, hd=64, S=512, B=16, FF=2048.
// ---------------------------------------------------------------------------

typedef __attribute__((ext_vector_type(16))) __bf16 v16bf;
typedef __attribute__((ext_vector_type(8)))  float  v8f;
typedef unsigned int v4u  __attribute__((ext_vector_type(4)));
typedef int          v8i_ __attribute__((ext_vector_type(8)));
typedef int          v4i_ __attribute__((ext_vector_type(4)));

union FragBF {
    v16bf    v;
    uint32_t u[8];
};

static __device__ __forceinline__ uint16_t f2bf(float f) {
    uint32_t u = __float_as_uint(f);
    uint32_t r = (u + 0x7FFFu + ((u >> 16) & 1u)) >> 16;   // RNE
    return (uint16_t)r;
}
static __device__ __forceinline__ uint32_t pack2bf(float lo, float hi) {
    return (uint32_t)f2bf(lo) | ((uint32_t)f2bf(hi) << 16);
}
static __device__ __forceinline__ v8f wmma_bf16(v16bf a, v16bf b, v8f c) {
    return __builtin_amdgcn_wmma_f32_16x16x32_bf16(false, a, false, b, (short)0, c,
                                                   false, false);
}

// ---------------------------------------------------------------------------
// TDM: DMA one 64x64 bf16 tile (row stride = strideElems) global -> LDS.
// D# per ISA 8.3-8.5: 2D tile, data_size=2B, dense LDS packing (8 KB).
// Issued once per wave (EXEC-ignored); tracked by TENSORcnt.
// clang-23 builtin: (u32x4 g0, i32x8 g1, i32x4 g2, i32x4 g3, i32x8, i32 cpol)
// ---------------------------------------------------------------------------
static __device__ __forceinline__ void tdm_load_64x64_bf16(
    uint32_t lds_byte_off, const uint16_t* gptr, int strideElems) {
    uint64_t ga = (uint64_t)(uintptr_t)gptr;
    v4u g0;
    g0[0] = 1u;                                         // count=1, user desc
    g0[1] = lds_byte_off;                               // lds_addr
    g0[2] = (uint32_t)ga;                               // global_addr[31:0]
    g0[3] = (uint32_t)((ga >> 32) & 0x01FFFFFFu) | (2u << 30);  // addr[56:32]|type=2
    v8i_ g1;
    g1[0] = (int)(1u << 16);                            // data_size=1 (2 bytes)
    g1[1] = (int)(64u << 16);                           // tensor_dim0 = 64
    g1[2] = (int)(64u << 16);                           // tensor_dim1 = 64
    g1[3] = (int)(64u << 16);                           // tile_dim0   = 64
    g1[4] = 64;                                         // tile_dim1=64, tile_dim2=0
    g1[5] = strideElems;                                // tensor_dim0_stride[31:0]
    g1[6] = 0;                                          // stride[47:32]=0, dim1_stride lo
    g1[7] = 0;
    v4i_ z4 = {0, 0, 0, 0};                             // groups 2/3 unused (2D)
    v8i_ z8 = {0, 0, 0, 0, 0, 0, 0, 0};                 // unused trailing operand
    __builtin_amdgcn_tensor_load_to_lds(g0, g1, z4, z4, z8, 0);
}
static __device__ __forceinline__ uint32_t lds_off32(const void* p) {
    return (uint32_t)(uintptr_t)p;   // low 32 bits of generic LDS addr = WG offset
}

// ---------------------------------------------------------------------------
// f32 -> bf16 bulk convert (weights), 4 elems/thread, packed dword stores.
// ---------------------------------------------------------------------------
__global__ __launch_bounds__(256) void cvt_bf16_kernel(const float* __restrict__ s,
                                                       uint16_t* __restrict__ d) {
    uint32_t* d32 = (uint32_t*)d;
    size_t p = ((size_t)blockIdx.x * 256 + threadIdx.x) * 2;   // dword pairs
    d32[p]     = pack2bf(s[p * 2],     s[p * 2 + 1]);
    d32[p + 1] = pack2bf(s[p * 2 + 2], s[p * 2 + 3]);
}

// ---------------------------------------------------------------------------
// LayerNorm: one wave per token (E=512, 16 elems/lane), shuffle reduction.
// BF16OUT=true writes bf16 (GEMM A operand), false writes f32 (final output).
// ---------------------------------------------------------------------------
template <bool BF16OUT>
__global__ __launch_bounds__(256) void ln_kernel(const float* __restrict__ x,
                                                 const float* __restrict__ w,
                                                 const float* __restrict__ b,
                                                 void* __restrict__ outp) {
    const int tok  = blockIdx.x * 8 + (threadIdx.x >> 5);
    const int lane = threadIdx.x & 31;
    const float* xp = x + (size_t)tok * 512;
    float vals[16];
    float s = 0.f, s2 = 0.f;
    for (int i = 0; i < 16; ++i) {
        float v = xp[lane + i * 32];
        vals[i] = v; s += v; s2 += v * v;
    }
    for (int m = 16; m >= 1; m >>= 1) {
        s  += __shfl_xor(s,  m, 32);
        s2 += __shfl_xor(s2, m, 32);
    }
    const float mu   = s * (1.f / 512.f);
    const float var  = s2 * (1.f / 512.f) - mu * mu;
    const float rinv = rsqrtf(var + 1e-5f);
    for (int i = 0; i < 16; ++i) {
        int e = lane + i * 32;
        float o = (vals[i] - mu) * rinv * w[e] + b[e];
        if (BF16OUT) ((uint16_t*)outp)[(size_t)tok * 512 + e] = f2bf(o);
        else         ((float*)outp)[(size_t)tok * 512 + e]   = o;
    }
}

// ---------------------------------------------------------------------------
// WMMA GEMM:  C[T=8192, N] = A[T,K](bf16) @ W[N,K](bf16)^T + bias, epilogues:
//   MODE 0: QKV  -> scatter bf16 q/k/v [B,H,S,hd], q *= 0.125
//   MODE 1: out-projection residual:  outF += gemm   (f32)
//   MODE 2: fc1 + ReLU:               outH  = relu(gemm)  (bf16)
//   MODE 3: fc2 residual:             outF += gemm   (f32)
// Block: 128 thr (4 waves), 64x64 tile, K-step 64, TDM double-buffered LDS.
// ---------------------------------------------------------------------------
template <int MODE>
__global__ __launch_bounds__(128) void gemm_kernel(
    const uint16_t* __restrict__ Aq, const uint16_t* __restrict__ Akv,
    const uint16_t* __restrict__ Wt, const float* __restrict__ bias,
    float* __restrict__ outF, uint16_t* __restrict__ outH,
    uint16_t* __restrict__ qo, uint16_t* __restrict__ ko, uint16_t* __restrict__ vo,
    int N, int K) {
    __shared__ uint32_t Abuf[2][64 * 32];   // 64 rows x 64 bf16 (8 KB) x2
    __shared__ uint32_t Wbuf[2][64 * 32];

    const int tid   = threadIdx.x;
    const int lane  = tid & 31;
    const int wave  = tid >> 5;
    const int wm    = (wave & 1) * 32;
    const int wn    = (wave >> 1) * 32;
    const int nTile = blockIdx.x * 64;
    const int tTile = blockIdx.y * 64;
    const int hi    = lane >> 4;
    const int l15   = lane & 15;

    const uint16_t* A = (MODE == 0 && nTile >= 512) ? Akv : Aq;  // cross-attn k/v

    const v8f zero = {0.f, 0.f, 0.f, 0.f, 0.f, 0.f, 0.f, 0.f};
    v8f acc[2][2];
    acc[0][0] = zero; acc[0][1] = zero; acc[1][0] = zero; acc[1][1] = zero;

    const int nStep = K >> 6;
    if (tid < 32) {   // wave 0 issues DMA for tile 0
        tdm_load_64x64_bf16(lds_off32(&Abuf[0][0]), A  + (size_t)tTile * K, K);
        tdm_load_64x64_bf16(lds_off32(&Wbuf[0][0]), Wt + (size_t)nTile * K, K);
    }
    for (int kt = 0; kt < nStep; ++kt) {
        const int p = kt & 1;
        if (kt + 1 < nStep) {           // prefetch next tile, wait for current
            const int k1 = (kt + 1) << 6;
            if (tid < 32) {
                tdm_load_64x64_bf16(lds_off32(&Abuf[p ^ 1][0]),
                                    A + (size_t)tTile * K + k1, K);
                tdm_load_64x64_bf16(lds_off32(&Wbuf[p ^ 1][0]),
                                    Wt + (size_t)nTile * K + k1, K);
                __builtin_amdgcn_s_wait_tensorcnt(2);
            }
        } else {
            if (tid < 32) __builtin_amdgcn_s_wait_tensorcnt(0);
        }
        __syncthreads();
        const uint32_t* As32 = &Abuf[p][0];
        const uint32_t* Ws32 = &Wbuf[p][0];
        for (int c = 0; c < 2; ++c) {   // two K=32 WMMA slabs per staged tile
            FragBF a[2], b[2];
            for (int mi = 0; mi < 2; ++mi) {
                int mloc = wm + mi * 16 + l15;
                for (int r = 0; r < 8; ++r) {
                    int kp = hi * 8 + r;   // A 16x32 layout
                    a[mi].u[r] = As32[mloc * 32 + c * 16 + kp];
                }
            }
            for (int ni = 0; ni < 2; ++ni) {
                int nloc = wn + ni * 16 + l15;
                for (int r = 0; r < 8; ++r)                         // B 32x16 layout
                    b[ni].u[r] = Ws32[nloc * 32 + c * 16 + hi * 8 + r];
            }
            for (int mi = 0; mi < 2; ++mi)
                for (int ni = 0; ni < 2; ++ni)
                    acc[mi][ni] = wmma_bf16(a[mi].v, b[ni].v, acc[mi][ni]);
        }
        __syncthreads();
    }

    // epilogue (D f32 16x16: element (m = r + 8*hi, n = lane&15))
    for (int mi = 0; mi < 2; ++mi)
        for (int ni = 0; ni < 2; ++ni)
            for (int r = 0; r < 8; ++r) {
                int mg = tTile + wm + mi * 16 + r + hi * 8;   // token
                int ng = nTile + wn + ni * 16 + l15;          // out feature
                float val = acc[mi][ni][r] + bias[ng];
                if (MODE == 0) {
                    int which = ng >> 9;        // 0=q 1=k 2=v
                    int e = ng & 511;
                    int h = e >> 6, d = e & 63;
                    int s = mg >> 4, bb = mg & 15;   // token = s*B + b, B=16
                    size_t off = (((size_t)(bb * 8 + h) * 512) + s) * 64 + d;
                    if (which == 0)      qo[off] = f2bf(val * 0.125f);
                    else if (which == 1) ko[off] = f2bf(val);
                    else                 vo[off] = f2bf(val);
                } else if (MODE == 2) {
                    outH[(size_t)mg * N + ng] = f2bf(val > 0.f ? val : 0.f);
                } else {   // MODE 1 / 3: residual accumulate in place (f32)
                    outF[(size_t)mg * N + ng] += val;
                }
            }
}

// ---------------------------------------------------------------------------
// Attention: one wave per (b, h, 16-query tile).  q,k,v bf16 [B,H,S,64].
// Scores via 64 WMMAs, softmax in LDS, P*V via 64 WMMAs, out bf16 [S,B,E].
// ---------------------------------------------------------------------------
__global__ __launch_bounds__(32) void attn_kernel(const uint16_t* __restrict__ q,
                                                  const uint16_t* __restrict__ k,
                                                  const uint16_t* __restrict__ v,
                                                  uint16_t* __restrict__ out) {
    __shared__ float sc[16][512];   // 32 KB score tile

    int idx = blockIdx.x;
    const int qt = idx & 31;  idx >>= 5;
    const int h  = idx & 7;
    const int b  = idx >> 3;
    const int lane = threadIdx.x;
    const int hi   = lane >> 4;
    const int l15  = lane & 15;
    const size_t bh = ((size_t)b * 8 + h) * 512;

    FragBF qa[2];
    {
        const uint32_t* qrow = (const uint32_t*)(q + (bh + (size_t)qt * 16 + l15) * 64);
        for (int c = 0; c < 2; ++c)
            for (int r = 0; r < 8; ++r) {
                int kp = hi * 8 + r;
                qa[c].u[r] = qrow[c * 16 + kp];
            }
    }
    for (int kt = 0; kt < 32; ++kt) {
        FragBF kb[2];
        const uint32_t* krow = (const uint32_t*)(k + (bh + (size_t)kt * 16 + l15) * 64);
        for (int c = 0; c < 2; ++c)
            for (int r = 0; r < 8; ++r)
                kb[c].u[r] = krow[c * 16 + hi * 8 + r];
        v8f s = {0.f, 0.f, 0.f, 0.f, 0.f, 0.f, 0.f, 0.f};
        s = wmma_bf16(qa[0].v, kb[0].v, s);
        s = wmma_bf16(qa[1].v, kb[1].v, s);
        for (int r = 0; r < 8; ++r)
            sc[r + hi * 8][kt * 16 + l15] = s[r];
    }
    __syncthreads();
    {   // softmax: lane owns row l15, half 'hi'; combine halves via shfl
        float* row = &sc[l15][hi * 256];
        float mx = -1e30f;
        for (int i = 0; i < 256; ++i) mx = fmaxf(mx, row[i]);
        mx = fmaxf(mx, __shfl_xor(mx, 16, 32));
        float sum = 0.f;
        for (int i = 0; i < 256; ++i) { float e = __expf(row[i] - mx); row[i] = e; sum += e; }
        sum += __shfl_xor(sum, 16, 32);
        float inv = 1.f / sum;
        for (int i = 0; i < 256; ++i) row[i] *= inv;
    }
    __syncthreads();
    const v8f zero = {0.f, 0.f, 0.f, 0.f, 0.f, 0.f, 0.f, 0.f};
    v8f o[4] = {zero, zero, zero, zero};
    for (int kc = 0; kc < 16; ++kc) {
        FragBF pa;
        for (int r = 0; r < 8; ++r) {
            int kk = hi * 16 + r * 2;
            const float* p = &sc[l15][kc * 32 + kk];
            pa.u[r] = pack2bf(p[0], p[1]);
        }
        for (int dt = 0; dt < 4; ++dt) {
            FragBF vb;
            int d = dt * 16 + l15;
            for (int r = 0; r < 8; ++r) {
                int kr = kc * 32 + hi * 16 + 2 * r;
                uint16_t lo = v[(bh + kr) * 64 + d];
                uint16_t hv = v[(bh + kr + 1) * 64 + d];
                vb.u[r] = (uint32_t)lo | ((uint32_t)hv << 16);
            }
            o[dt] = wmma_bf16(pa.v, vb.v, o[dt]);
        }
    }
    for (int dt = 0; dt < 4; ++dt)
        for (int r = 0; r < 8; ++r) {
            int s = qt * 16 + r + hi * 8;
            int d = dt * 16 + l15;
            out[((size_t)s * 16 + b) * 512 + h * 64 + d] = f2bf(o[dt][r]);
        }
}

// ---------------------------------------------------------------------------
// Host orchestration
// ---------------------------------------------------------------------------
extern "C" void kernel_launch(void* const* d_in, const int* in_sizes, int n_in,
                              void* d_out, int out_size, void* d_ws, size_t ws_size,
                              hipStream_t stream) {
    (void)in_sizes; (void)n_in; (void)out_size; (void)ws_size;
    constexpr int E = 512, S = 512, FF = 2048;
    constexpr size_t TOK = (size_t)S * 16;             // 8192 tokens per stream
    constexpr size_t XB  = TOK * E * sizeof(float);
    constexpr size_t HB  = TOK * E * sizeof(uint16_t);

    const float* ln1w = (const float*)d_in[2];  const float* ln1b = (const float*)d_in[3];
    const float* siW  = (const float*)d_in[4];  const float* siB  = (const float*)d_in[5];
    const float* soW  = (const float*)d_in[6];  const float* soB  = (const float*)d_in[7];
    const float* ln2w = (const float*)d_in[8];  const float* ln2b = (const float*)d_in[9];
    const float* ciW  = (const float*)d_in[10]; const float* ciB  = (const float*)d_in[11];
    const float* coW  = (const float*)d_in[12]; const float* coB  = (const float*)d_in[13];
    const float* ln3w = (const float*)d_in[14]; const float* ln3b = (const float*)d_in[15];
    const float* f1W  = (const float*)d_in[16]; const float* f1B  = (const float*)d_in[17];
    const float* f2W  = (const float*)d_in[18]; const float* f2B  = (const float*)d_in[19];
    const float* lnfw = (const float*)d_in[20]; const float* lnfb = (const float*)d_in[21];

    char* cur = (char*)d_ws;
    auto carve = [&](size_t bytes) { char* p = cur; cur += (bytes + 255) & ~(size_t)255; return p; };
    float*    xb[2]  = { (float*)carve(XB), (float*)carve(XB) };
    uint16_t* xn[2]  = { (uint16_t*)carve(HB), (uint16_t*)carve(HB) };
    uint16_t* qb[2]  = { (uint16_t*)carve(HB), (uint16_t*)carve(HB) };
    uint16_t* kb[2]  = { (uint16_t*)carve(HB), (uint16_t*)carve(HB) };
    uint16_t* vb[2]  = { (uint16_t*)carve(HB), (uint16_t*)carve(HB) };
    uint16_t* at[2]  = { (uint16_t*)carve(HB), (uint16_t*)carve(HB) };
    uint16_t* ffh    = (uint16_t*)carve(TOK * FF * sizeof(uint16_t));
    // bf16 weight caches
    const size_t nSI = 2 * 2 * 3 * (size_t)E * E, nSO = 2 * 2 * (size_t)E * E;
    const size_t nF1 = 2 * 2 * (size_t)FF * E;
    uint16_t* siWb = (uint16_t*)carve(nSI * 2);
    uint16_t* soWb = (uint16_t*)carve(nSO * 2);
    uint16_t* ciWb = (uint16_t*)carve(nSI * 2);
    uint16_t* coWb = (uint16_t*)carve(nSO * 2);
    uint16_t* f1Wb = (uint16_t*)carve(nF1 * 2);
    uint16_t* f2Wb = (uint16_t*)carve(nF1 * 2);

    (void)hipMemcpyAsync(xb[0], d_in[0], XB, hipMemcpyDeviceToDevice, stream);
    (void)hipMemcpyAsync(xb[1], d_in[1], XB, hipMemcpyDeviceToDevice, stream);
    cvt_bf16_kernel<<<nSI / 1024, 256, 0, stream>>>(siW, siWb);
    cvt_bf16_kernel<<<nSO / 1024, 256, 0, stream>>>(soW, soWb);
    cvt_bf16_kernel<<<nSI / 1024, 256, 0, stream>>>(ciW, ciWb);
    cvt_bf16_kernel<<<nSO / 1024, 256, 0, stream>>>(coW, coWb);
    cvt_bf16_kernel<<<nF1 / 1024, 256, 0, stream>>>(f1W, f1Wb);
    cvt_bf16_kernel<<<nF1 / 1024, 256, 0, stream>>>(f2W, f2Wb);

    const dim3 gQKV(24, 128), gPROJ(8, 128), gFC1(32, 128), gFC2(8, 128);

    for (int l = 0; l < 2; ++l) {
        // ---- self attention ----
        for (int m = 0; m < 2; ++m) {
            size_t o = (size_t)(l * 2 + m) * E;
            ln_kernel<true><<<TOK / 8, 256, 0, stream>>>(xb[m], ln1w + o, ln1b + o, xn[m]);
        }
        for (int m = 0; m < 2; ++m) {
            size_t i = (size_t)(l * 2 + m);
            gemm_kernel<0><<<gQKV, 128, 0, stream>>>(xn[m], xn[m],
                siWb + i * 3 * E * E, siB + i * 3 * E, nullptr, nullptr,
                qb[m], kb[m], vb[m], 3 * E, E);
        }
        for (int m = 0; m < 2; ++m)
            attn_kernel<<<4096, 32, 0, stream>>>(qb[m], kb[m], vb[m], at[m]);
        for (int m = 0; m < 2; ++m) {
            size_t i = (size_t)(l * 2 + m);
            gemm_kernel<1><<<gPROJ, 128, 0, stream>>>(at[m], at[m],
                soWb + i * E * E, soB + i * E, xb[m], nullptr,
                nullptr, nullptr, nullptr, E, E);
        }
        // ---- cross attention (q from self stream, k/v from other) ----
        for (int m = 0; m < 2; ++m) {
            size_t o = (size_t)(l * 2 + m) * E;
            ln_kernel<true><<<TOK / 8, 256, 0, stream>>>(xb[m], ln2w + o, ln2b + o, xn[m]);
        }
        for (int m = 0; m < 2; ++m) {
            size_t i = (size_t)(l * 2 + m);
            gemm_kernel<0><<<gQKV, 128, 0, stream>>>(xn[m], xn[1 - m],
                ciWb + i * 3 * E * E, ciB + i * 3 * E, nullptr, nullptr,
                qb[m], kb[m], vb[m], 3 * E, E);
        }
        for (int m = 0; m < 2; ++m)
            attn_kernel<<<4096, 32, 0, stream>>>(qb[m], kb[m], vb[m], at[m]);
        for (int m = 0; m < 2; ++m) {
            size_t i = (size_t)(l * 2 + m);
            gemm_kernel<1><<<gPROJ, 128, 0, stream>>>(at[m], at[m],
                coWb + i * E * E, coB + i * E, xb[m], nullptr,
                nullptr, nullptr, nullptr, E, E);
        }
        // ---- FFN ----
        for (int m = 0; m < 2; ++m) {
            size_t o = (size_t)(l * 2 + m) * E;
            ln_kernel<true><<<TOK / 8, 256, 0, stream>>>(xb[m], ln3w + o, ln3b + o, xn[m]);
        }
        for (int m = 0; m < 2; ++m) {   // sequential per-stream: ffh reused
            size_t i = (size_t)(l * 2 + m);
            gemm_kernel<2><<<gFC1, 128, 0, stream>>>(xn[m], xn[m],
                f1Wb + i * FF * E, f1B + i * FF, nullptr, ffh,
                nullptr, nullptr, nullptr, FF, E);
            gemm_kernel<3><<<gFC2, 128, 0, stream>>>(ffh, ffh,
                f2Wb + i * E * FF, f2B + i * E, xb[m], nullptr,
                nullptr, nullptr, nullptr, E, FF);
        }
    }
    // ---- final layernorm -> d_out [M,S,B,E] f32 ----
    for (int m = 0; m < 2; ++m)
        ln_kernel<false><<<TOK / 8, 256, 0, stream>>>(xb[m], lnfw + (size_t)m * E,
                                                      lnfb + (size_t)m * E,
                                                      (float*)d_out + (size_t)m * TOK * E);
}
